// ScannedRNN_42906723287521
// MI455X (gfx1250) — matrix operand order repaired and checked by reading, not compiled, not run on hardware
//
#include <hip/hip_runtime.h>

// ---------------------------------------------------------------------------
// Scanned GRU for MI455X (gfx1250, wave32).
//   Phase 0: weights -> bf16 [n][k]; x -> bf16 (async-to-LDS source).
//   Phase 1: gi = x @ Wi + bi  — bf16 WMMA GEMM (f32 accum), A tile staged
//            with GLOBAL_LOAD_ASYNC_TO_LDS_B128 (ASYNCcnt path).
//   Phase 2: sequential scan; h @ Wh via bf16 WMMA, batch split into 4
//            independent row groups x 8 column-split WGs synced by a
//            per-group global barrier each timestep (ping-pong h buffer,
//            bf16 shadow staged async into LDS each step).
// Inner loops deliberately left in simple load->WMMA form: the gfx1250
// scheduler produces copy-free, hazard-free WMMA streams from it, and
// occupancy (2+ waves/SIMD) hides the L2-resident B-fragment latency.
// ---------------------------------------------------------------------------

typedef __bf16 bf16_t;
typedef __attribute__((ext_vector_type(16))) __bf16 v16bf;
typedef __attribute__((ext_vector_type(8)))  __bf16 v8bf;
typedef __attribute__((ext_vector_type(8)))  float  v8f;

#define T_STEPS 512
#define B_DIM   64
#define E_DIM   1024
#define G3E     3072   // 3*E

// Workspace layout (bytes)
#define WS_WIT   ((size_t)0)                        // bf16 [3072][1024]
#define WS_WHT   ((size_t)6291456)                  // bf16 [3072][1024]
#define WS_XBF   ((size_t)12582912)                 // bf16 [T*B][1024]
#define WS_GI    (WS_XBF + (size_t)67108864)        // f32  [T][B][3E]
#define WS_HBUF  (WS_GI + (size_t)402653184)        // f32  [2][B][E]
#define WS_HBF   (WS_HBUF + (size_t)524288)         // bf16 [2][B][E]
#define WS_BARS  (WS_HBF + (size_t)262144)          // int  [4][2]

__device__ __forceinline__ float sigmoid_f(float x) {
    return 1.0f / (1.0f + __expf(-x));
}
__device__ __forceinline__ float tanh_f(float x) {
    float e = __expf(-2.0f * x);
    return (1.0f - e) / (1.0f + e);
}

// Wave-relative LDS byte offset: low 32 bits of the generic shared-var address
// (LDS aperture places the allocation offset in addr[31:0]).
__device__ __forceinline__ unsigned lds_off_u32(const void* p) {
    return (unsigned)(unsigned long long)p;
}

// Per-lane async copy of 16 bytes global -> LDS (tracked on ASYNCcnt).
__device__ __forceinline__ void async_copy_b128(unsigned lds_addr, const void* gaddr) {
    asm volatile("global_load_async_to_lds_b128 %0, %1, off"
                 :: "v"(lds_addr), "v"(gaddr) : "memory");
}
__device__ __forceinline__ void wait_asynccnt0() {
    asm volatile("s_wait_asynccnt 0x0" ::: "memory");
}

// Sense/generation global barrier over `members` workgroups (agent scope).
__device__ __forceinline__ void group_barrier(int* bar, int members) {
    __syncthreads();
    if (threadIdx.x == 0) {
        __threadfence();  // make h writes visible at device scope
        int gen  = __hip_atomic_load(bar + 1, __ATOMIC_ACQUIRE, __HIP_MEMORY_SCOPE_AGENT);
        int prev = __hip_atomic_fetch_add(bar, 1, __ATOMIC_ACQ_REL, __HIP_MEMORY_SCOPE_AGENT);
        if (prev == members - 1) {
            __hip_atomic_store(bar, 0, __ATOMIC_RELAXED, __HIP_MEMORY_SCOPE_AGENT);
            __hip_atomic_fetch_add(bar + 1, 1, __ATOMIC_ACQ_REL, __HIP_MEMORY_SCOPE_AGENT);
        } else {
            while (__hip_atomic_load(bar + 1, __ATOMIC_ACQUIRE, __HIP_MEMORY_SCOPE_AGENT) == gen) {
                __builtin_amdgcn_s_sleep(8);
            }
        }
    }
    __syncthreads();
}

// ---------------------------------------------------------------------------
// Kernel 0a: transpose + convert Wi, Wh (f32 [k][n]) -> bf16 [n][k].
// ---------------------------------------------------------------------------
__global__ void cvt_weights(const float* __restrict__ Wi,
                            const float* __restrict__ Wh,
                            bf16_t* __restrict__ WiT,
                            bf16_t* __restrict__ WhT) {
    long i = (long)blockIdx.x * 256 + threadIdx.x;   // over 3072*1024
    int n = (int)(i >> 10);
    int k = (int)(i & 1023);
    if (n < G3E) {
        WiT[i] = (bf16_t)Wi[(long)k * G3E + n];
        WhT[i] = (bf16_t)Wh[(long)k * G3E + n];
    }
}

// ---------------------------------------------------------------------------
// Kernel 0b: x (f32) -> bf16, contiguous, 4 elements per thread.
// ---------------------------------------------------------------------------
__global__ void cvt_x(const float* __restrict__ x, bf16_t* __restrict__ xbf) {
    long i = ((long)blockIdx.x * 256 + threadIdx.x) * 4;
    float4 v = *(const float4*)(x + i);
    union { bf16_t b[4]; unsigned long long u; } o;
    o.b[0] = (bf16_t)v.x; o.b[1] = (bf16_t)v.y;
    o.b[2] = (bf16_t)v.z; o.b[3] = (bf16_t)v.w;
    *(unsigned long long*)(xbf + i) = o.u;
}

// ---------------------------------------------------------------------------
// Kernel 1: gi = x @ Wi + bi.  grid = (TB/16, 3072/512), block = 256 (8 waves).
// A tile (16x1024 bf16) staged with async-to-LDS; each wave owns 4 N-tiles.
// ---------------------------------------------------------------------------
__global__ void __launch_bounds__(256) gemm_gi(const bf16_t* __restrict__ xbf,
                                               const bf16_t* __restrict__ WiT,
                                               const float* __restrict__ bi,
                                               float* __restrict__ gi) {
    __shared__ bf16_t Ah[16][1032];   // row stride 2064 B: 16B-aligned, conflict-free
    const int r0   = blockIdx.x * 16;        // row block within TB = 32768
    const int nblk = blockIdx.y * 512;       // output-column block base

    // Async-stage xbf[r0..r0+15][0..1023] -> LDS (2048 x 16B chunks).
    const bf16_t* xrow = xbf + (long)r0 * E_DIM;
    for (int chunk = threadIdx.x; chunk < 2048; chunk += 256) {
        int r   = chunk >> 7;      // 128 chunks of 16B per 1024-col row
        int c16 = chunk & 127;
        async_copy_b128(lds_off_u32(&Ah[r][c16 * 8]),
                        xrow + (long)r * E_DIM + c16 * 8);
    }
    wait_asynccnt0();
    __syncthreads();

    const int lane = threadIdx.x & 31;
    const int wave = threadIdx.x >> 5;
    const int m    = lane & 15;     // A row / B column / C column within tile
    const int half = lane >> 4;     // lane half selects K sub-range
    const int c0   = nblk + wave * 64;

    v8f acc[4] = {v8f{}, v8f{}, v8f{}, v8f{}};

#pragma unroll 4
    for (int k0 = 0; k0 < E_DIM; k0 += 32) {
        // A fragment (16x32): lanes<16 K{0-15},
        // lanes>=16 K{16-31}; contiguous 16B chunks from LDS.
        union { v16bf v; v8bf h[2]; } a;
        a.h[0] = *(const v8bf*)&Ah[m][k0 + half * 16];
        a.h[1] = *(const v8bf*)&Ah[m][k0 + half * 16 + 8];
#pragma unroll
        for (int t = 0; t < 4; ++t) {
            // B fragment (32x16): lane n = tile col, 16 contiguous K values.
            const bf16_t* bp = WiT + (long)(c0 + t * 16 + m) * E_DIM + k0 + half * 16;
            v16bf b = *(const v16bf*)bp;
            acc[t] = __builtin_amdgcn_wmma_f32_16x16x32_bf16(
                false, a.v, false, b, (short)0, acc[t], false, false);
        }
    }

    // Store C (+bias). C layout: col = lane&15, rows = (lane>>4)*8 + v.
#pragma unroll
    for (int t = 0; t < 4; ++t) {
        int c = c0 + t * 16 + m;
        float bias = bi[c];
#pragma unroll
        for (int v = 0; v < 8; ++v) {
            int r = half * 8 + v;
            gi[(long)(r0 + r) * G3E + c] = acc[t][v] + bias;
        }
    }
}

// ---------------------------------------------------------------------------
// Kernel init: zero the per-row-group barriers (fresh state every launch).
// ---------------------------------------------------------------------------
__global__ void init_bars(int* bars) {
    if (threadIdx.x < 8) bars[threadIdx.x] = 0;
}

// ---------------------------------------------------------------------------
// Kernel 2: the scan.  grid = 32 WGs = 4 row groups (16 batch rows) x 8
// column groups (128 h columns). Each wave owns 16 h columns -> 3 gate
// accumulator tiles (r/z/n). f32 carry in ping-pong buffer; bf16 shadow
// staged async into LDS each step; done flags staged to LDS once per step.
// ---------------------------------------------------------------------------
__global__ void __launch_bounds__(256) scan_gru(const float* __restrict__ gi,
                                                const bf16_t* __restrict__ WhT,
                                                const float* __restrict__ bhn,
                                                const float* __restrict__ h0,
                                                const int* __restrict__ dones,
                                                float* __restrict__ ys,
                                                float* __restrict__ hbuf,
                                                bf16_t* __restrict__ hbf,
                                                int* bars) {
    __shared__ bf16_t Ah[16][1032];
    __shared__ int done_s[16];
    const int rg = blockIdx.x >> 3;          // row group 0..3
    const int cg = blockIdx.x & 7;           // column group 0..7
    const int b0 = rg * 16;
    const int lane = threadIdx.x & 31;
    const int wave = threadIdx.x >> 5;
    const int m    = lane & 15;
    const int half = lane >> 4;
    const int e0   = cg * 128 + wave * 16;   // wave's h-column base
    int* bar = bars + rg * 2;
    float*  hb[2]  = { hbuf, hbuf + (size_t)B_DIM * E_DIM };
    bf16_t* hbb[2] = { hbf,  hbf  + (size_t)B_DIM * E_DIM };

    // B base pointers for the three gates (r/z/n rows of WhT).
    const bf16_t* pgate[3];
#pragma unroll
    for (int g = 0; g < 3; ++g)
        pgate[g] = WhT + (long)(g * E_DIM + e0 + m) * E_DIM + half * 16;

    // Initialize carry (f32 + bf16 shadow) from h0 slice.
    for (int idx = threadIdx.x; idx < 16 * 128; idx += 256) {
        int r = idx >> 7, c = (idx & 127) + cg * 128;
        float v = h0[(long)(b0 + r) * E_DIM + c];
        hb[0][(long)(b0 + r) * E_DIM + c]  = v;
        hbb[0][(long)(b0 + r) * E_DIM + c] = (bf16_t)v;
    }
    group_barrier(bar, 8);

    int cur = 0;
    for (int t = 0; t < T_STEPS; ++t) {
        const float*  hsrc  = hb[cur];
        float*        hdst  = hb[cur ^ 1];
        const bf16_t* hbsrc = hbb[cur];
        bf16_t*       hbdst = hbb[cur ^ 1];

        // Async-stage bf16 h rows b0..b0+15 (full 1024 cols) -> LDS.
        const bf16_t* hrow = hbsrc + (long)b0 * E_DIM;
        for (int chunk = threadIdx.x; chunk < 2048; chunk += 256) {
            int r   = chunk >> 7;
            int c16 = chunk & 127;
            async_copy_b128(lds_off_u32(&Ah[r][c16 * 8]),
                            hrow + (long)r * E_DIM + c16 * 8);
        }
        // Stage per-row done flags while the async copies are in flight.
        if (threadIdx.x < 16) done_s[threadIdx.x] = dones[t * B_DIM + b0 + threadIdx.x];
        wait_asynccnt0();
        __syncthreads();

        // Apply done-mask: zero whole LDS rows for done batch entries.
        for (int r = 0; r < 16; ++r) {
            if (done_s[r]) {
                for (int c = threadIdx.x; c < E_DIM; c += 256) Ah[r][c] = (bf16_t)0.0f;
            }
        }
        __syncthreads();

        v8f accr = v8f{}, accz = v8f{}, accn = v8f{};
#pragma unroll 4
        for (int k0 = 0; k0 < E_DIM; k0 += 32) {
            union { v16bf v; v8bf h[2]; } a;
            a.h[0] = *(const v8bf*)&Ah[m][k0 + half * 16];
            a.h[1] = *(const v8bf*)&Ah[m][k0 + half * 16 + 8];
            v16bf br_ = *(const v16bf*)(pgate[0] + k0);
            v16bf bz_ = *(const v16bf*)(pgate[1] + k0);
            v16bf bn_ = *(const v16bf*)(pgate[2] + k0);
            accr = __builtin_amdgcn_wmma_f32_16x16x32_bf16(false, a.v, false, br_, (short)0, accr, false, false);
            accz = __builtin_amdgcn_wmma_f32_16x16x32_bf16(false, a.v, false, bz_, (short)0, accz, false, false);
            accn = __builtin_amdgcn_wmma_f32_16x16x32_bf16(false, a.v, false, bn_, (short)0, accn, false, false);
        }

        // Fused gating. C layout: col = lane&15 -> e, rows = half*8 + v.
        const int   e  = e0 + m;
        const float bn = bhn[e];
        const float* git = gi + (long)t * B_DIM * G3E;
#pragma unroll
        for (int v = 0; v < 8; ++v) {
            int   br    = b0 + half * 8 + v;
            float hprev = done_s[half * 8 + v] ? 0.0f
                                               : hsrc[(long)br * E_DIM + e];  // f32 masked carry
            const float* gr = git + (long)br * G3E;
            float r  = sigmoid_f(gr[e]             + accr[v]);
            float z  = sigmoid_f(gr[E_DIM + e]     + accz[v]);
            float n  = tanh_f  (gr[2 * E_DIM + e]  + r * (accn[v] + bn));
            float hn = (1.0f - z) * n + z * hprev;
            hdst[(long)br * E_DIM + e]  = hn;
            hbdst[(long)br * E_DIM + e] = (bf16_t)hn;
            ys[((long)t * B_DIM + br) * E_DIM + e] = hn;
        }

        group_barrier(bar, 8);   // also orders LDS re-stage for next step
        cur ^= 1;
    }
}

// ---------------------------------------------------------------------------
extern "C" void kernel_launch(void* const* d_in, const int* in_sizes, int n_in,
                              void* d_out, int out_size, void* d_ws, size_t ws_size,
                              hipStream_t stream) {
    const float* x    = (const float*)d_in[0];   // (T,B,E)
    const int*   dn   = (const int*)  d_in[1];   // (T,B)
    const float* h0   = (const float*)d_in[2];   // (B,E)
    const float* Wi   = (const float*)d_in[3];   // (E,3E)
    const float* Wh   = (const float*)d_in[4];   // (E,3E)
    const float* bi   = (const float*)d_in[5];   // (3E,)
    const float* bhn  = (const float*)d_in[6];   // (E,)
    float*       ys   = (float*)d_out;           // (T,B,E)

    char* ws = (char*)d_ws;
    bf16_t* WiT  = (bf16_t*)(ws + WS_WIT);
    bf16_t* WhT  = (bf16_t*)(ws + WS_WHT);
    bf16_t* xbf  = (bf16_t*)(ws + WS_XBF);
    float*  gi   = (float*) (ws + WS_GI);
    float*  hbuf = (float*) (ws + WS_HBUF);
    bf16_t* hbf  = (bf16_t*)(ws + WS_HBF);
    int*    bars = (int*)   (ws + WS_BARS);

    // Phase 0: weights -> bf16 [n][k]; x -> bf16.
    cvt_weights<<<(G3E * E_DIM) / 256, 256, 0, stream>>>(Wi, Wh, WiT, WhT);
    cvt_x<<<(T_STEPS * B_DIM * E_DIM) / (256 * 4), 256, 0, stream>>>(x, xbf);

    // Phase 1: gi = x @ Wi + bi over all T*B rows.
    gemm_gi<<<dim3((T_STEPS * B_DIM) / 16, G3E / 512), 256, 0, stream>>>(xbf, WiT, bi, gi);

    // Phase 2: sequential scan (32 persistent WGs, per-row-group barriers).
    init_bars<<<1, 32, 0, stream>>>(bars);
    scan_gru<<<32, 256, 0, stream>>>(gi, WhT, bhn, h0, dn, ys, hbuf, hbf, bars);
}
